// DiffAttn_31138512896418
// MI455X (gfx1250) — matrix operand order repaired and checked by reading, not compiled, not run on hardware
//
#include <hip/hip_runtime.h>
#include <hip/hip_bf16.h>

// ---------------------------------------------------------------------------
// Types for CDNA5 WMMA (wave32): v_wmma_f32_16x16x32_bf16
// ---------------------------------------------------------------------------
typedef __attribute__((ext_vector_type(16))) __bf16 v16bf;
typedef __attribute__((ext_vector_type(8)))  float  v8f;
typedef __attribute__((ext_vector_type(4)))  unsigned int u32x4;
typedef __attribute__((ext_vector_type(4)))  int          i32x4;

union frag_u { v16bf v; u32x4 q[2]; };

// CDNA5 async global->LDS copies (ASYNCcnt path), guarded so a toolchain
// without the builtins still compiles via the synchronous fallback.
#if defined(__gfx1250__) && defined(__has_builtin)
#if __has_builtin(__builtin_amdgcn_global_load_async_to_lds_b128) && \
    __has_builtin(__builtin_amdgcn_s_wait_asynccnt)
#define HAVE_ASYNC_LDS 1
#endif
#endif
#ifndef HAVE_ASYNC_LDS
#define HAVE_ASYNC_LDS 0
#endif

__device__ __forceinline__ __bf16 f2bf(float f) {
    union { float f; unsigned int u; } a; a.f = f;
    unsigned int r = a.u + 0x7FFFu + ((a.u >> 16) & 1u);   // RNE
    unsigned short h = (unsigned short)(r >> 16);
    union { unsigned short s; __bf16 b; } o; o.s = h;
    return o.b;
}

// 16B global -> LDS copy, async when the CDNA5 builtin is available.
// Builtin prototype (from compiler diagnostic): (v4i addrspace(1)*, v4i addrspace(3)*, imm, imm)
__device__ __forceinline__ void cp16(const __bf16* __restrict__ gsrc,
                                     __bf16* __restrict__ ldst) {
#if HAVE_ASYNC_LDS
    __builtin_amdgcn_global_load_async_to_lds_b128(
        (__attribute__((address_space(1))) i32x4*)gsrc,
        (__attribute__((address_space(3))) i32x4*)ldst, 0, 0);
#else
    *(u32x4*)ldst = *(const u32x4*)gsrc;
#endif
}
__device__ __forceinline__ void cp16_wait() {
#if HAVE_ASYNC_LDS
    __builtin_amdgcn_s_wait_asynccnt(0);
#endif
}

// A-fragment style: two 16B chunks (K 0..7 / 8..15 for lanes 0-15, 16..23 / 24..31 for 16-31)
__device__ __forceinline__ v16bf load_frag2(const __bf16* p0, const __bf16* p1) {
    frag_u r;
    r.q[0] = *(const u32x4*)p0;
    r.q[1] = *(const u32x4*)p1;
    return r.v;
}
// B-fragment style: one contiguous 32B chunk (16 consecutive K for this lane's column)
__device__ __forceinline__ v16bf load_fragc(const __bf16* p) {
    frag_u r;
    r.q[0] = *(const u32x4*)p;
    r.q[1] = *(const u32x4*)(p + 8);
    return r.v;
}

__device__ __forceinline__ v8f wmma_bf16(v16bf a, v16bf b, v8f c) {
    return __builtin_amdgcn_wmma_f32_16x16x32_bf16(false, a, false, b, (short)0, c,
                                                   false, false);
}

#define LAMBDA_INIT      0.7836057665f
#define ONE_MINUS_LINIT  0.2163942335f

// ---------------------------------------------------------------------------
// fp32 -> bf16 conversion (activations)
// ---------------------------------------------------------------------------
__global__ void cvt_f32_bf16(const float* __restrict__ in, __bf16* __restrict__ out, int n) {
    int i = blockIdx.x * blockDim.x + threadIdx.x;
    if (i < n) out[i] = f2bf(in[i]);
}

// ---------------------------------------------------------------------------
// fp32 -> bf16 transposed conversion for 1024x1024 weights: Wt[n][k] = W[k][n].
// Transposing once here means GEMM B-tiles are contiguous (no LDS scatter).
// ---------------------------------------------------------------------------
__global__ __launch_bounds__(256) void cvt_t_bf16(const float* __restrict__ in,
                                                  __bf16* __restrict__ out) {
    __shared__ float tile[32][33];
    const int k0 = blockIdx.y * 32, n0 = blockIdx.x * 32;
    const int tx = threadIdx.x, ty = threadIdx.y;       // 32 x 8
#pragma unroll
    for (int i = 0; i < 32; i += 8)
        tile[ty + i][tx] = in[(size_t)(k0 + ty + i) * 1024 + n0 + tx];
    __syncthreads();
#pragma unroll
    for (int i = 0; i < 32; i += 8)
        out[(size_t)(n0 + ty + i) * 1024 + k0 + tx] = f2bf(tile[tx][ty + i]);
}

// ---------------------------------------------------------------------------
// lambda_full = exp(sum lq1*lk1) - exp(sum lq2*lk2) + LAMBDA_INIT
// ---------------------------------------------------------------------------
__global__ void lambda_kernel(const float* __restrict__ lq1, const float* __restrict__ lk1,
                              const float* __restrict__ lq2, const float* __restrict__ lk2,
                              float* __restrict__ lam) {
    if (threadIdx.x == 0) {
        float s1 = 0.f, s2 = 0.f;
        for (int i = 0; i < 64; ++i) { s1 += lq1[i] * lk1[i]; s2 += lq2[i] * lk2[i]; }
        *lam = __expf(s1) - __expf(s2) + LAMBDA_INIT;
    }
}

// ---------------------------------------------------------------------------
// Tiled bf16 GEMM: C(4096x1024) = A(4096x1024) x Wt^T  (Wt is N x K), fp32 accum.
// Block tile 128x128, K-step 32, 8 waves each doing 2x4 WMMA 16x16 tiles.
// Both LDS tiles are staged with contiguous 16B (async) copies.
// Epilogue modes:
//   0: Q  -> bf16 (B,2H,S,64), scaled by D^-0.5
//   1: K  -> bf16 (B,2H,S,64)
//   2: V  -> bf16 transposed (B,H,128,S)   (so PV B-fragments are contiguous)
//   3: out-> fp32 row-major (4096x1024)
// ---------------------------------------------------------------------------
__global__ __launch_bounds__(256) void gemm_bf16(const __bf16* __restrict__ A,
                                                 const __bf16* __restrict__ Wt,
                                                 void* __restrict__ outp,
                                                 int mode, float scale) {
    constexpr int GK = 1024, GN = 1024;
    __shared__ alignas(16) __bf16 As[128 * 32];   // [m][k]
    __shared__ alignas(16) __bf16 Bs[128 * 32];   // [n][k]

    const int tid  = threadIdx.x;
    const int m0   = blockIdx.y * 128;
    const int n0   = blockIdx.x * 128;
    const int w    = tid >> 5;
    const int lane = tid & 31;
    const int lo   = lane & 15, hi = lane >> 4;
    const int wm   = (w >> 1) * 32;   // wave M offset: 0/32/64/96
    const int wn   = (w & 1) * 64;    // wave N offset: 0/64

    v8f acc[2][4];
#pragma unroll
    for (int i = 0; i < 2; ++i)
#pragma unroll
        for (int j = 0; j < 4; ++j)
#pragma unroll
            for (int e = 0; e < 8; ++e) acc[i][j][e] = 0.f;

    const int sr = tid >> 1;            // staged row 0..127 (shared by A and B tiles)
    const int sc = (tid & 1) * 16;      // 16-elem column group

    const __bf16* gA = A  + (size_t)(m0 + sr) * GK + sc;
    const __bf16* gB = Wt + (size_t)(n0 + sr) * GK + sc;
    __bf16* lA = As + sr * 32 + sc;
    __bf16* lB = Bs + sr * 32 + sc;

    for (int k0 = 0; k0 < GK; k0 += 32) {
        __syncthreads();                 // previous iteration's reads are done
        cp16(gA + k0,     lA);
        cp16(gA + k0 + 8, lA + 8);
        cp16(gB + k0,     lB);
        cp16(gB + k0 + 8, lB + 8);
        cp16_wait();
        if (k0 + 32 < GK) {              // prefetch next K tiles into cache
            __builtin_prefetch(gA + k0 + 32, 0, 3);
            __builtin_prefetch(gB + k0 + 32, 0, 3);
        }
        __syncthreads();

        v16bf af[2];
#pragma unroll
        for (int mt = 0; mt < 2; ++mt) {
            const __bf16* p = As + (wm + mt * 16 + lo) * 32;
            af[mt] = load_frag2(p + hi * 16, p + hi * 16 + 8);
        }
#pragma unroll
        for (int nt = 0; nt < 4; ++nt) {
            const __bf16* p = Bs + (wn + nt * 16 + lo) * 32 + hi * 16;
            v16bf bfr = load_fragc(p);
#pragma unroll
            for (int mt = 0; mt < 2; ++mt)
                acc[mt][nt] = wmma_bf16(af[mt], bfr, acc[mt][nt]);
        }
    }

    // epilogue: C layout -> lane lo is N, VGPR r is M = r + 8*hi
#pragma unroll
    for (int mt = 0; mt < 2; ++mt) {
#pragma unroll
        for (int nt = 0; nt < 4; ++nt) {
#pragma unroll
            for (int r = 0; r < 8; ++r) {
                const int row = m0 + wm + mt * 16 + r + hi * 8;  // 0..4095
                const int col = n0 + wn + nt * 16 + lo;          // 0..1023
                const float v = acc[mt][nt][r] * scale;
                const int b = row >> 11, s = row & 2047;
                if (mode == 0 || mode == 1) {
                    const int h2 = col >> 6, d = col & 63;
                    ((__bf16*)outp)[((size_t)(b * 16 + h2) * 2048 + s) * 64 + d] = f2bf(v);
                } else if (mode == 2) {
                    const int h = col >> 7, dd = col & 127;
                    ((__bf16*)outp)[((size_t)(b * 8 + h) * 128 + dd) * 2048 + s] = f2bf(v);
                } else {
                    ((float*)outp)[(size_t)row * GN + col] = v;
                }
            }
        }
    }
}

// ---------------------------------------------------------------------------
// Flash differential attention body: for each (b, h2) virtual head,
// O[b,h2] = softmax(Q K^T) V  with online softmax, bf16 WMMA, fp32 state.
// Block = 128 threads (4 waves), each wave owns 16 query rows.
// ---------------------------------------------------------------------------
__global__ __launch_bounds__(128) void diff_attn_fwd(const __bf16* __restrict__ Q,   // (B,16,S,64), pre-scaled
                                                     const __bf16* __restrict__ K,   // (B,16,S,64)
                                                     const __bf16* __restrict__ Vt,  // (B,8,128,S)
                                                     float* __restrict__ O) {        // (B,16,S,128)
    constexpr int S = 2048;
    __shared__ alignas(16) __bf16 Plds[4][16 * 64];

    const int w    = threadIdx.x >> 5;
    const int lane = threadIdx.x & 31;
    const int lo   = lane & 15, hi = lane >> 4;
    const int q0   = blockIdx.x * 64 + w * 16;
    const int h2   = blockIdx.y;
    const int b    = blockIdx.z;

    const __bf16* Qb = Q + (size_t)(b * 16 + h2) * S * 64;
    const __bf16* Kb = K + (size_t)(b * 16 + h2) * S * 64;
    const __bf16* Vb = Vt + (size_t)(b * 8 + (h2 >> 1)) * 128 * S;

    // Q A-fragments for this wave's 16 rows (K-steps over d=0..31, 32..63)
    v16bf qf[2];
#pragma unroll
    for (int ks = 0; ks < 2; ++ks) {
        const __bf16* p = Qb + (size_t)(q0 + lo) * 64 + ks * 32;
        qf[ks] = load_frag2(p + hi * 16, p + hi * 16 + 8);
    }

    v8f o[8];
    float m_i[8], l_i[8];
#pragma unroll
    for (int nt = 0; nt < 8; ++nt)
#pragma unroll
        for (int e = 0; e < 8; ++e) o[nt][e] = 0.f;
#pragma unroll
    for (int r = 0; r < 8; ++r) { m_i[r] = -1e30f; l_i[r] = 0.f; }

    for (int kv0 = 0; kv0 < S; kv0 += 64) {
        if (kv0 + 64 < S) {     // cache prefetch of next KV block
            __builtin_prefetch(Kb + (size_t)(kv0 + 64 + lo * 4) * 64, 0, 3);
            __builtin_prefetch(Vb + (size_t)(lane * 4) * S + kv0 + 64, 0, 3);
        }
        // ---- S tile (16x64) = Q (16x64) x K^T ----
        v8f sa[4];
#pragma unroll
        for (int nt = 0; nt < 4; ++nt) {
#pragma unroll
            for (int e = 0; e < 8; ++e) sa[nt][e] = 0.f;
#pragma unroll
            for (int ks = 0; ks < 2; ++ks) {
                // B-fragment: lane's column = kv row kv0+nt*16+lo, contiguous d
                const __bf16* p = Kb + (size_t)(kv0 + nt * 16 + lo) * 64 + ks * 32 + hi * 16;
                v16bf kf = load_fragc(p);
                sa[nt] = wmma_bf16(qf[ks], kf, sa[nt]);
            }
        }

        // ---- online softmax (row = r + 8*hi, spread over 16 lanes) ----
#pragma unroll
        for (int r = 0; r < 8; ++r) {
            float rm = fmaxf(fmaxf(sa[0][r], sa[1][r]), fmaxf(sa[2][r], sa[3][r]));
            rm = fmaxf(rm, __shfl_xor(rm, 1, 32));
            rm = fmaxf(rm, __shfl_xor(rm, 2, 32));
            rm = fmaxf(rm, __shfl_xor(rm, 4, 32));
            rm = fmaxf(rm, __shfl_xor(rm, 8, 32));
            const float mn   = fmaxf(m_i[r], rm);
            const float corr = __expf(m_i[r] - mn);
            m_i[r] = mn;
            l_i[r] *= corr;
#pragma unroll
            for (int nt2 = 0; nt2 < 8; ++nt2) o[nt2][r] *= corr;
            float rs = 0.f;
#pragma unroll
            for (int nt = 0; nt < 4; ++nt) {
                const float pv = __expf(sa[nt][r] - mn);
                sa[nt][r] = pv;
                rs += pv;
            }
            rs += __shfl_xor(rs, 1, 32);
            rs += __shfl_xor(rs, 2, 32);
            rs += __shfl_xor(rs, 4, 32);
            rs += __shfl_xor(rs, 8, 32);
            l_i[r] += rs;
        }

        // ---- P (C-layout) -> LDS bf16 -> A-layout fragments ----
#pragma unroll
        for (int nt = 0; nt < 4; ++nt)
#pragma unroll
            for (int r = 0; r < 8; ++r)
                Plds[w][(r + hi * 8) * 64 + nt * 16 + lo] = f2bf(sa[nt][r]);

        v16bf pf[2];
#pragma unroll
        for (int ks = 0; ks < 2; ++ks) {
            const __bf16* p = &Plds[w][lo * 64 + ks * 32];
            pf[ks] = load_frag2(p + hi * 16, p + hi * 16 + 8);
        }

        // ---- O (16x128) += P (16x64) x V (64x128), V^T gives contiguous B-frags
#pragma unroll
        for (int nt2 = 0; nt2 < 8; ++nt2) {
#pragma unroll
            for (int ks = 0; ks < 2; ++ks) {
                const __bf16* p = Vb + (size_t)(nt2 * 16 + lo) * S + kv0 + ks * 32 + hi * 16;
                v16bf vf = load_fragc(p);
                o[nt2] = wmma_bf16(pf[ks], vf, o[nt2]);
            }
        }
    }

    // ---- epilogue: normalize by l and store fp32 ----
    float* Ob = O + (size_t)(b * 16 + h2) * S * 128;
#pragma unroll
    for (int r = 0; r < 8; ++r) {
        const float inv = 1.0f / l_i[r];
        const int row = q0 + r + hi * 8;
#pragma unroll
        for (int nt2 = 0; nt2 < 8; ++nt2)
            Ob[(size_t)row * 128 + nt2 * 16 + lo] = o[nt2][r] * inv;
    }
}

// ---------------------------------------------------------------------------
// Combine head pairs + RMSNorm + affine + (1 - lambda_init); out bf16 (B,S,1024)
// One block = one (b,h,s) row of 128.
// ---------------------------------------------------------------------------
__global__ __launch_bounds__(128) void combine_rms(const float* __restrict__ O,   // (B,16,S,128)
                                                   const float* __restrict__ lam,
                                                   const float* __restrict__ subw,
                                                   __bf16* __restrict__ out) {     // (B,S,1024)
    constexpr int S = 2048;
    __shared__ float red[4];
    const int blk = blockIdx.x;          // b*8*S + h*S + s
    const int s = blk & 2047;
    const int h = (blk >> 11) & 7;
    const int b = blk >> 14;
    const int dd = threadIdx.x;
    const float lamf = *lam;

    const size_t base0 = ((size_t)(b * 16 + 2 * h) * S + s) * 128;
    const size_t base1 = ((size_t)(b * 16 + 2 * h + 1) * S + s) * 128;
    const float v = O[base0 + dd] - lamf * O[base1 + dd];

    float ss = v * v;
    ss += __shfl_xor(ss, 1, 32);
    ss += __shfl_xor(ss, 2, 32);
    ss += __shfl_xor(ss, 4, 32);
    ss += __shfl_xor(ss, 8, 32);
    ss += __shfl_xor(ss, 16, 32);
    if ((threadIdx.x & 31) == 0) red[threadIdx.x >> 5] = ss;
    __syncthreads();
    const float tot = red[0] + red[1] + red[2] + red[3];
    const float rms = rsqrtf(tot * (1.0f / 128.0f) + 1e-5f);

    const float ov = v * rms * subw[dd] * ONE_MINUS_LINIT;
    out[((size_t)(b * 2048 + s) * 1024) + h * 128 + dd] = f2bf(ov);
}

// ---------------------------------------------------------------------------
// Launch
// ---------------------------------------------------------------------------
extern "C" void kernel_launch(void* const* d_in, const int* in_sizes, int n_in,
                              void* d_out, int out_size, void* d_ws, size_t ws_size,
                              hipStream_t stream) {
    const float* query = (const float*)d_in[0];  // (2,2048,1024)
    const float* Wq    = (const float*)d_in[1];
    const float* Wk    = (const float*)d_in[2];
    const float* Wv    = (const float*)d_in[3];
    const float* Wo    = (const float*)d_in[4];
    const float* lq1   = (const float*)d_in[5];
    const float* lk1   = (const float*)d_in[6];
    const float* lq2   = (const float*)d_in[7];
    const float* lk2   = (const float*)d_in[8];
    const float* subw  = (const float*)d_in[9];

    char* ws = (char*)d_ws;
    __bf16* Xbf  = (__bf16*)(ws + 0);               // 8 MB  (4096x1024)
    __bf16* Wqt  = (__bf16*)(ws + 8388608);         // 2 MB  (transposed N x K)
    __bf16* Wkt  = (__bf16*)(ws + 10485760);        // 2 MB
    __bf16* Wvt  = (__bf16*)(ws + 12582912);        // 2 MB
    __bf16* Wot  = (__bf16*)(ws + 14680064);        // 2 MB
    __bf16* Qbf  = (__bf16*)(ws + 16777216);        // 8 MB (B,16,S,64)
    __bf16* Kbf  = (__bf16*)(ws + 25165824);        // 8 MB (B,16,S,64)
    __bf16* Vtb  = (__bf16*)(ws + 33554432);        // 8 MB (B,8,128,S)
    float*  Obuf = (float*)(ws + 41943040);         // 32 MB (B,16,S,128)
    __bf16* Attn = (__bf16*)(ws + 75497472);        // 8 MB (B,S,1024)
    float*  lam  = (float*)(ws + 83886080);         // 4 B

    // 1) convert activations; transpose-convert weights (Wt[n][k])
    cvt_f32_bf16<<<(4194304 + 255) / 256, 256, 0, stream>>>(query, Xbf, 4194304);
    dim3 tb(32, 8), tg(32, 32);
    cvt_t_bf16<<<tg, tb, 0, stream>>>(Wq, Wqt);
    cvt_t_bf16<<<tg, tb, 0, stream>>>(Wk, Wkt);
    cvt_t_bf16<<<tg, tb, 0, stream>>>(Wv, Wvt);
    cvt_t_bf16<<<tg, tb, 0, stream>>>(Wo, Wot);

    // 2) lambda_full
    lambda_kernel<<<1, 32, 0, stream>>>(lq1, lk1, lq2, lk2, lam);

    // 3) projections (WMMA GEMMs); Q pre-scaled by D^-0.5 = 0.125
    dim3 gg(1024 / 128, 4096 / 128);
    gemm_bf16<<<gg, 256, 0, stream>>>(Xbf, Wqt, (void*)Qbf, 0, 0.125f);
    gemm_bf16<<<gg, 256, 0, stream>>>(Xbf, Wkt, (void*)Kbf, 1, 1.0f);
    gemm_bf16<<<gg, 256, 0, stream>>>(Xbf, Wvt, (void*)Vtb, 2, 1.0f);

    // 4) flash attention for all 16 virtual heads
    diff_attn_fwd<<<dim3(2048 / 64, 16, 2), 128, 0, stream>>>(Qbf, Kbf, Vtb, Obuf);

    // 5) differential combine + RMSNorm
    combine_rms<<<2 * 8 * 2048, 128, 0, stream>>>(Obuf, lam, subw, Attn);

    // 6) output projection -> fp32 d_out
    gemm_bf16<<<gg, 256, 0, stream>>>(Attn, Wot, d_out, 3, 1.0f);
}
